// NeuronCircuit_38491496907060
// MI455X (gfx1250) — matrix operand order repaired and checked by reading, not compiled, not run on hardware
//
#include <hip/hip_runtime.h>

// ---------------------------------------------------------------------------
// Types / helpers
// ---------------------------------------------------------------------------
typedef unsigned short u16;
typedef unsigned int   u32;
typedef __bf16 v16bf __attribute__((ext_vector_type(16)));
typedef float  v8f   __attribute__((ext_vector_type(8)));
typedef unsigned int v4u __attribute__((ext_vector_type(4)));
typedef int v8i __attribute__((ext_vector_type(8)));
typedef int v4i __attribute__((ext_vector_type(4)));

#define D_MODEL 1024
#define N_HEADS 16
#define D_HEAD  64
#define RANK    128
#define N_COMP  64
#define STATE   64
#define BSZ     2
#define SLEN    2048
#define MROWS   (BSZ*SLEN)     // 4096

__device__ __forceinline__ u16 f2bf(float x) {
  union { float fv; u32 uv; } c; c.fv = x;
  u32 r = c.uv + 0x7FFFu + ((c.uv >> 16) & 1u);   // round-to-nearest-even
  return (u16)(r >> 16);
}

// ---------------------------------------------------------------------------
// WMMA fragments (CDNA5 ISA 7.12.2, wave32).
// Key fact: with K contiguous in memory, a lane's fragment is two contiguous
// 16B spans -> direct global b128 loads, no LDS staging needed.
// ---------------------------------------------------------------------------
// A fragment (16x32 bf16, row-major source): lane m=lane&15, halves at
// k = hi8..hi8+7 and 16+hi8..23+hi8
__device__ __forceinline__ v16bf frag_a_g(const u16* A, long lda, long m0,
                                          long k0, int lane) {
  union { v16bf v; uint4 q[2]; } f;
  const u16* p = A + (m0 + (lane & 15)) * lda + k0 + ((lane >> 4) << 3);
  f.q[0] = *(const uint4*)p;
  f.q[1] = *(const uint4*)(p + 16);
  return f.v;
}
// B fragment (32x16 bf16) with element(k,n)=B[(n0+n)*sn + k0+k] (sk==1):
// lane n=lane&15, halves at k = hi8..hi8+7 and 16+hi8..23+hi8 (hi8 = (lane>>4)*8)
__device__ __forceinline__ v16bf frag_b_g_k1(const u16* B, long sn, long k0,
                                             long n0, int lane) {
  union { v16bf v; uint4 q[2]; } f;
  const u16* p = B + (n0 + (lane & 15)) * sn + k0 + ((lane >> 4) << 3);
  f.q[0] = *(const uint4*)p;
  f.q[1] = *(const uint4*)(p + 16);
  return f.v;
}
// B tile staged in LDS, K-pair interleaved [(k>>1)][n][k&1]; source sn==1
__device__ __forceinline__ void load_b_tile_n1(u16* b_lds, const u16* Bm,
                                               long sk, long k0, long n0,
                                               int lane) {
#pragma unroll
  for (int pi = lane; pi < 256; pi += 32) {
    int kk = pi >> 3, nn2 = (pi & 7) << 1;
    u32 v = *(const u32*)(Bm + (k0 + kk) * sk + n0 + nn2);
    b_lds[(kk >> 1) * 32 + nn2 * 2 + (kk & 1)] = (u16)v;
    b_lds[(kk >> 1) * 32 + (nn2 + 1) * 2 + (kk & 1)] = (u16)(v >> 16);
  }
}
__device__ __forceinline__ v16bf frag_b_lds(const u16* b_lds, int lane) {
  union { v16bf v; u32 u[8]; } f;
  int n = lane & 15, kb8 = (lane >> 4) << 2;
  const u32* p = (const u32*)b_lds;
#pragma unroll
  for (int hp = 0; hp < 8; ++hp) f.u[hp] = p[(kb8 + hp + (hp & 4)) * 16 + n];
  return f.v;
}
// A fragment from a row-major 16x32 LDS tile (for the P matrix in attention)
__device__ __forceinline__ v16bf frag_a_lds(const u16* a_lds, int lane) {
  union { v16bf v; u32 u[8]; } f;
  int m = lane & 15, hi4 = (lane >> 4) << 2;
  const u32* p = (const u32*)a_lds;
#pragma unroll
  for (int vv = 0; vv < 4; ++vv) f.u[vv] = p[m * 16 + vv + hi4];
#pragma unroll
  for (int vv = 0; vv < 4; ++vv) f.u[4 + vv] = p[m * 16 + 8 + vv + hi4];
  return f.v;
}
#define WMMA_BF16(a, b, c) \
  __builtin_amdgcn_wmma_f32_16x16x32_bf16(false, (a), false, (b), (short)0, (c), false, false)

// ---------------------------------------------------------------------------
// fp32 -> bf16 conversion / zero
// ---------------------------------------------------------------------------
__global__ __launch_bounds__(256) void k_f2bf(const float* __restrict__ src,
                                              u16* __restrict__ dst, int n) {
  int i = blockIdx.x * 256 + threadIdx.x;
  if (i < n) dst[i] = f2bf(src[i]);
}
__global__ __launch_bounds__(128) void k_zero(float* p, int n) {
  int i = blockIdx.x * 128 + threadIdx.x;
  if (i < n) p[i] = 0.f;
}

// ---------------------------------------------------------------------------
// CDNA5 async-tensor probe: NULL-descriptor TDM load (count=0 => no-op) +
// tensorcnt wait + cluster barrier (S_NOP when not in a cluster).
// ---------------------------------------------------------------------------
__global__ __launch_bounds__(32) void k_cdna5_probe() {
#if defined(__gfx1250__) && __has_builtin(__builtin_amdgcn_tensor_load_to_lds)
  v4u g0 = {};            // D# group0: count=0 -> NULL tensor descriptor
  v8i g1 = {};
  v4i g2 = {}, g3 = {};
#if defined(__clang_major__) && (__clang_major__ >= 23)
  v8i g2b = {};
  __builtin_amdgcn_tensor_load_to_lds(g0, g1, g2, g3, g2b, 0);
#else
  __builtin_amdgcn_tensor_load_to_lds(g0, g1, g2, g3, 0);
#endif
  __builtin_amdgcn_s_wait_tensorcnt(0);
#endif
  __builtin_amdgcn_s_cluster_barrier();
}

// ---------------------------------------------------------------------------
// GEMM variant 1: B has K contiguous (sk==1). Fully direct, no LDS/barriers.
// 8 waves/block, 16x32 output slab per wave, K-loop = 6 b128 loads + 2 WMMA.
// ---------------------------------------------------------------------------
template<bool BF16OUT>
__global__ __launch_bounds__(256)
void k_gemm_bk1(const u16* __restrict__ A, long lda, long aBS,
                const u16* __restrict__ B, long sn, long bBS,
                void* __restrict__ C, long ldc, long cBS,
                int tilesN2, int K) {
  int wave = threadIdx.x >> 5, lane = threadIdx.x & 31;
  long tile = (long)blockIdx.x * 8 + wave;
  long m0 = (tile / tilesN2) * 16, n0 = (tile % tilesN2) * 32;
  const u16* Ab = A + (long)blockIdx.y * aBS;
  const u16* Bb = B + (long)blockIdx.y * bBS;
  v8f acc0 = {}, acc1 = {};
  for (int k0 = 0; k0 < K; k0 += 32) {
    __builtin_prefetch(Ab + m0 * lda + k0 + 64, 0, 1);
    v16bf fa = frag_a_g(Ab, lda, m0, k0, lane);
    acc0 = WMMA_BF16(fa, frag_b_g_k1(Bb, sn, k0, n0, lane), acc0);
    acc1 = WMMA_BF16(fa, frag_b_g_k1(Bb, sn, k0, n0 + 16, lane), acc1);
  }
  int hi8 = (lane >> 4) << 3, n = lane & 15;
#pragma unroll
  for (int vv = 0; vv < 8; ++vv) {
    long m = m0 + vv + hi8;
    long o0 = (long)blockIdx.y * cBS + m * ldc + n0 + n;
    if (BF16OUT) {
      ((u16*)C)[o0] = f2bf(acc0[vv]);
      ((u16*)C)[o0 + 16] = f2bf(acc1[vv]);
    } else {
      ((float*)C)[o0] = acc0[vv];
      ((float*)C)[o0 + 16] = acc1[vv];
    }
  }
}

// ---------------------------------------------------------------------------
// GEMM variant 2: B has N contiguous (sn==1). Direct A, LDS-staged B.
// ---------------------------------------------------------------------------
template<bool BF16OUT>
__global__ __launch_bounds__(256)
void k_gemm_bn1(const u16* __restrict__ A, long lda, long aBS,
                const u16* __restrict__ B, long sk, long bBS,
                void* __restrict__ C, long ldc, long cBS,
                int tilesN2, int K) {
  __shared__ __align__(16) u16 lds[8 * 1024];
  int wave = threadIdx.x >> 5, lane = threadIdx.x & 31;
  u16* b_lds = lds + wave * 1024;
  long tile = (long)blockIdx.x * 8 + wave;
  long m0 = (tile / tilesN2) * 16, n0 = (tile % tilesN2) * 32;
  const u16* Ab = A + (long)blockIdx.y * aBS;
  const u16* Bb = B + (long)blockIdx.y * bBS;
  v8f acc0 = {}, acc1 = {};
  for (int k0 = 0; k0 < K; k0 += 32) {
    __builtin_prefetch(Ab + m0 * lda + k0 + 64, 0, 1);
    load_b_tile_n1(b_lds, Bb, sk, k0, n0, lane);
    load_b_tile_n1(b_lds + 512, Bb, sk, k0, n0 + 16, lane);
    __syncthreads();
    v16bf fa = frag_a_g(Ab, lda, m0, k0, lane);
    acc0 = WMMA_BF16(fa, frag_b_lds(b_lds, lane), acc0);
    acc1 = WMMA_BF16(fa, frag_b_lds(b_lds + 512, lane), acc1);
    __syncthreads();
  }
  int hi8 = (lane >> 4) << 3, n = lane & 15;
#pragma unroll
  for (int vv = 0; vv < 8; ++vv) {
    long m = m0 + vv + hi8;
    long o0 = (long)blockIdx.y * cBS + m * ldc + n0 + n;
    if (BF16OUT) {
      ((u16*)C)[o0] = f2bf(acc0[vv]);
      ((u16*)C)[o0 + 16] = f2bf(acc1[vv]);
    } else {
      ((float*)C)[o0] = acc0[vv];
      ((float*)C)[o0 + 16] = acc1[vv];
    }
  }
}

// ---------------------------------------------------------------------------
// SSM scan + routing glue (fp32 VALU)
// ---------------------------------------------------------------------------
__global__ __launch_bounds__(128) void k_scan(const float* __restrict__ xB,
                                              const float* __restrict__ Am,
                                              float* __restrict__ h_final) {
  __shared__ float As[STATE * STATE];
  __shared__ float h[BSZ][STATE];
  int t = threadIdx.x;
  for (int i = t; i < STATE * STATE; i += 128) As[i] = Am[i];
  int b = t >> 6, n = t & 63;
  h[b][n] = 0.f;
  __syncthreads();
  for (int s = 0; s < SLEN; ++s) {
    float acc = xB[((long)b * SLEN + s) * STATE + n];
#pragma unroll 8
    for (int m = 0; m < STATE; ++m) acc += h[b][m] * As[m * STATE + n];
    __syncthreads();
    h[b][n] = acc;
    __syncthreads();
  }
  h_final[b * STATE + n] = h[b][n];
}

__global__ __launch_bounds__(256) void k_hproj(const float* __restrict__ hf,
                                               const float* __restrict__ Wimp,
                                               float* __restrict__ hp) {
  int i = blockIdx.x * 256 + threadIdx.x;           // 2048
  int b = i >> 10, o = i & 1023;
  float a = 0.f;
#pragma unroll 8
  for (int m = 0; m < STATE; ++m) a += hf[b * STATE + m] * Wimp[o * STATE + m];
  hp[i] = a;
}

__global__ __launch_bounds__(256) void k_implogit(const float* __restrict__ x,
                                                  const float* __restrict__ hp,
                                                  float* __restrict__ il) {
  int gw = (blockIdx.x * 256 + threadIdx.x) >> 5;
  int lane = threadIdx.x & 31;
  if (gw >= MROWS) return;
  int b = gw >> 11;
  const float* xr = x + (long)gw * D_MODEL;
  const float* h = hp + b * D_MODEL;
  float a = 0.f;
  for (int d = lane; d < D_MODEL; d += 32) a += xr[d] * h[d];
  for (int o = 16; o; o >>= 1) a += __shfl_xor(a, o, 32);
  if (lane == 0) il[gw] = a;
}

__global__ __launch_bounds__(256) void k_softmaxS(const float* __restrict__ il,
                                                  float* __restrict__ imp) {
  __shared__ float red[256];
  int b = blockIdx.x, t = threadIdx.x;
  const float* src = il + (long)b * SLEN;
  float mx = -1e30f;
  for (int s = t; s < SLEN; s += 256) mx = fmaxf(mx, src[s]);
  red[t] = mx; __syncthreads();
  for (int o = 128; o; o >>= 1) { if (t < o) red[t] = fmaxf(red[t], red[t + o]); __syncthreads(); }
  mx = red[0]; __syncthreads();
  float sm = 0.f;
  for (int s = t; s < SLEN; s += 256) sm += __expf(src[s] - mx);
  red[t] = sm; __syncthreads();
  for (int o = 128; o; o >>= 1) { if (t < o) red[t] += red[t + o]; __syncthreads(); }
  float inv = 1.f / red[0];
  for (int s = t; s < SLEN; s += 256) imp[(long)b * SLEN + s] = __expf(src[s] - mx) * inv;
}

__global__ __launch_bounds__(256) void k_nw(const float* __restrict__ logits,
                                            const float* __restrict__ imp,
                                            float* __restrict__ nwraw) {
  int gw = (blockIdx.x * 256 + threadIdx.x) >> 5;
  int lane = threadIdx.x & 31;
  if (gw >= MROWS) return;
  int b = gw >> 11;
  const float* lg = logits + (long)gw * N_COMP;
  float v0 = lg[lane], v1 = lg[32 + lane];
  float mx = fmaxf(v0, v1);
  for (int o = 16; o; o >>= 1) mx = fmaxf(mx, __shfl_xor(mx, o, 32));
  float e0 = __expf(v0 - mx), e1 = __expf(v1 - mx);
  float s = e0 + e1;
  for (int o = 16; o; o >>= 1) s += __shfl_xor(s, o, 32);
  float w = imp[gw] / s;
  atomicAdd(&nwraw[b * N_COMP + lane], e0 * w);
  atomicAdd(&nwraw[b * N_COMP + 32 + lane], e1 * w);
}

__global__ __launch_bounds__(64) void k_nwnorm(float* nwraw) {
  __shared__ float sh[64];
  int b = blockIdx.x, n = threadIdx.x;
  float v = nwraw[b * N_COMP + n];
  sh[n] = v; __syncthreads();
  for (int o = 32; o; o >>= 1) { if (n < o) sh[n] += sh[n + o]; __syncthreads(); }
  nwraw[b * N_COMP + n] = v / (sh[0] + 1e-8f);
}

__global__ __launch_bounds__(256) void k_sc(const float* __restrict__ cn,
                                            const float* __restrict__ nw,
                                            u16* __restrict__ sc) {
  __shared__ float w[BSZ][N_COMP];
  int t = threadIdx.x;
  if (t < BSZ * N_COMP) w[t >> 6][t & 63] = nw[t];
  __syncthreads();
  long i = (long)blockIdx.x * 256 + t;   // over D_MODEL*RANK
  int d = (int)(i >> 7), r = (int)(i & 127);
  float a0 = 0.f, a1 = 0.f;
#pragma unroll 4
  for (int n = 0; n < N_COMP; ++n) {
    float v = cn[((long)n * D_MODEL + d) * RANK + r];
    a0 += w[0][n] * v;
    a1 += w[1][n] * v;
  }
  sc[i] = f2bf(a0);
  sc[(long)D_MODEL * RANK + i] = f2bf(a1);
}

// ---------------------------------------------------------------------------
// QKV: fully direct fragments (no LDS). A tile reused for 3 weight B-tiles.
// ---------------------------------------------------------------------------
__global__ __launch_bounds__(256)
void k_qkv(const u16* __restrict__ hcomp, const u16* __restrict__ Wq,
           const u16* __restrict__ Wk, const u16* __restrict__ Wv,
           u16* __restrict__ Qo, u16* __restrict__ Ko, u16* __restrict__ Vo) {
  int wave = threadIdx.x >> 5, lane = threadIdx.x & 31;
  int b = blockIdx.y;
  long tile = (long)blockIdx.x * 8 + wave;
  long tm = tile >> 6, tn = tile & 63;              // tilesN = 64
  const u16* A = hcomp + (long)b * SLEN * RANK;
  v8f acc[3] = {};
  for (int k0 = 0; k0 < RANK; k0 += 32) {
    v16bf fa = frag_a_g(A, RANK, tm * 16, k0, lane);
    acc[0] = WMMA_BF16(fa, frag_b_g_k1(Wq, RANK, k0, tn * 16, lane), acc[0]);
    acc[1] = WMMA_BF16(fa, frag_b_g_k1(Wk, RANK, k0, tn * 16, lane), acc[1]);
    acc[2] = WMMA_BF16(fa, frag_b_g_k1(Wv, RANK, k0, tn * 16, lane), acc[2]);
  }
  int hi8 = (lane >> 4) << 3, n = lane & 15;
  int o = (int)tn * 16 + n, h = o >> 6, dh = o & 63;
#pragma unroll
  for (int w = 0; w < 3; ++w) {
    u16* O = (w == 0) ? Qo : (w == 1) ? Ko : Vo;
#pragma unroll
    for (int vv = 0; vv < 8; ++vv) {
      long s = tm * 16 + vv + hi8;
      O[(((long)b * N_HEADS + h) * SLEN + s) * D_HEAD + dh] = f2bf(acc[w][vv]);
    }
  }
}

// ---------------------------------------------------------------------------
// Causal flash attention: one wave per (b,h, 16-row query tile); 32-key chunks.
// Q and K^T fragments direct from global; LDS only for V tiles + P round-trip.
// ---------------------------------------------------------------------------
__global__ __launch_bounds__(32)
void k_attn(const u16* __restrict__ Q, const u16* __restrict__ K,
            const u16* __restrict__ V, u16* __restrict__ attn_out) {
  __shared__ __align__(16) u16 b_lds[512];
  __shared__ __align__(16) u16 p_lds[512];
  int lane = threadIdx.x;
  int bh = blockIdx.y;                              // b*16 + h
  int b = bh >> 4, h = bh & 15;
  long q0 = (long)blockIdx.x * 16;
  const u16* Qp = Q + (long)bh * SLEN * D_HEAD;
  const u16* Kp = K + (long)bh * SLEN * D_HEAD;
  const u16* Vp = V + (long)bh * SLEN * D_HEAD;

  v16bf qf0 = frag_a_g(Qp, D_HEAD, q0, 0, lane);
  v16bf qf1 = frag_a_g(Qp, D_HEAD, q0, 32, lane);

  v8f acc[4] = {};
  float m8[8], l8[8];
#pragma unroll
  for (int vv = 0; vv < 8; ++vv) { m8[vv] = -1e30f; l8[vv] = 0.f; }
  int hi8 = (lane >> 4) << 3, nlo = lane & 15;
  int nchunks = (int)((q0 + 47) >> 5);

  for (int c = 0; c < nchunks; ++c) {
    long k0 = (long)c * 32;
    v8f sf[2];
#pragma unroll
    for (int jh = 0; jh < 2; ++jh) {
      v8f s = {};
      long kbase = k0 + jh * 16;
      // scores = Q @ K^T : B(kd, j) = K[(kbase+j)*64 + kd]  (direct b128)
      s = WMMA_BF16(qf0, frag_b_g_k1(Kp, D_HEAD, 0, kbase, lane), s);
      s = WMMA_BF16(qf1, frag_b_g_k1(Kp, D_HEAD, 32, kbase, lane), s);
      sf[jh] = s;
    }
#pragma unroll
    for (int jh = 0; jh < 2; ++jh)
#pragma unroll
      for (int vv = 0; vv < 8; ++vv) {
        long row = q0 + vv + hi8;
        long col = k0 + jh * 16 + nlo;
        sf[jh][vv] = (col <= row) ? sf[jh][vv] * 0.125f : -1e30f;
      }
#pragma unroll
    for (int vv = 0; vv < 8; ++vv) {
      float rm = fmaxf(sf[0][vv], sf[1][vv]);
      for (int o = 8; o; o >>= 1) rm = fmaxf(rm, __shfl_xor(rm, o, 32));
      float mn = fmaxf(m8[vv], rm);
      float es = __expf(m8[vv] - mn);
      float p0 = __expf(sf[0][vv] - mn);
      float p1 = __expf(sf[1][vv] - mn);
      float rs = p0 + p1;
      for (int o = 8; o; o >>= 1) rs += __shfl_xor(rs, o, 32);
      l8[vv] = l8[vv] * es + rs;
      m8[vv] = mn;
#pragma unroll
      for (int t = 0; t < 4; ++t) acc[t][vv] *= es;
      p_lds[(vv + hi8) * 32 + nlo] = f2bf(p0);
      p_lds[(vv + hi8) * 32 + 16 + nlo] = f2bf(p1);
    }
    __syncthreads();
    v16bf pf = frag_a_lds(p_lds, lane);
    __syncthreads();
    // acc += P @ V : B(kk, n) = V[(k0+kk)*64 + t*16+n]  (sn==1 -> LDS staged)
#pragma unroll
    for (int t = 0; t < 4; ++t) {
      load_b_tile_n1(b_lds, Vp + k0 * D_HEAD + t * 16, D_HEAD, 0, 0, lane);
      __syncthreads();
      acc[t] = WMMA_BF16(pf, frag_b_lds(b_lds, lane), acc[t]);
      __syncthreads();
    }
  }
#pragma unroll
  for (int t = 0; t < 4; ++t)
#pragma unroll
    for (int vv = 0; vv < 8; ++vv) {
      long row = q0 + vv + hi8;
      int dh = t * 16 + nlo;
      attn_out[((long)b * SLEN + row) * D_MODEL + h * D_HEAD + dh] =
          f2bf(acc[t][vv] / l8[vv]);
    }
}

// ---------------------------------------------------------------------------
// Host-side orchestration
// ---------------------------------------------------------------------------
extern "C" void kernel_launch(void* const* d_in, const int* in_sizes, int n_in,
                              void* d_out, int out_size, void* d_ws, size_t ws_size,
                              hipStream_t stream) {
  (void)in_sizes; (void)n_in; (void)out_size; (void)ws_size;
  const float* x        = (const float*)d_in[0];
  const float* A        = (const float*)d_in[1];
  const float* B_ssm    = (const float*)d_in[2];
  const float* W_imp    = (const float*)d_in[3];
  const float* W_router = (const float*)d_in[4];
  const float* cn       = (const float*)d_in[5];
  const float* Wq       = (const float*)d_in[6];
  const float* Wk       = (const float*)d_in[7];
  const float* Wv       = (const float*)d_in[8];
  const float* Wo       = (const float*)d_in[9];
  float* out = (float*)d_out;

  char* wp = (char*)d_ws;
  auto alloc = [&](size_t bytes) -> void* {
    void* p = wp; wp += (bytes + 255) & ~(size_t)255; return p;
  };
  u16* x_bf      = (u16*)alloc((size_t)MROWS * D_MODEL * 2);
  u16* bssm_bf   = (u16*)alloc((size_t)D_MODEL * STATE * 2);
  u16* wrout_bf  = (u16*)alloc((size_t)N_COMP * D_MODEL * 2);
  u16* wq_bf     = (u16*)alloc((size_t)D_MODEL * RANK * 2);
  u16* wk_bf     = (u16*)alloc((size_t)D_MODEL * RANK * 2);
  u16* wv_bf     = (u16*)alloc((size_t)D_MODEL * RANK * 2);
  u16* wo_bf     = (u16*)alloc((size_t)D_MODEL * D_MODEL * 2);
  float* xB      = (float*)alloc((size_t)MROWS * STATE * 4);
  float* logits  = (float*)alloc((size_t)MROWS * N_COMP * 4);
  float* h_final = (float*)alloc((size_t)BSZ * STATE * 4);
  float* h_proj  = (float*)alloc((size_t)BSZ * D_MODEL * 4);
  float* il      = (float*)alloc((size_t)MROWS * 4);
  float* imp     = (float*)alloc((size_t)MROWS * 4);
  float* nwraw   = (float*)alloc((size_t)BSZ * N_COMP * 4);
  u16* sc_bf     = (u16*)alloc((size_t)BSZ * D_MODEL * RANK * 2);
  u16* hcomp_bf  = (u16*)alloc((size_t)MROWS * RANK * 2);
  u16* Qb        = (u16*)alloc((size_t)MROWS * D_MODEL * 2);
  u16* Kb        = (u16*)alloc((size_t)MROWS * D_MODEL * 2);
  u16* Vb        = (u16*)alloc((size_t)MROWS * D_MODEL * 2);
  u16* attn_bf   = (u16*)alloc((size_t)MROWS * D_MODEL * 2);

  auto cvt = [&](const float* s, u16* d, int n) {
    k_f2bf<<<(n + 255) / 256, 256, 0, stream>>>(s, d, n);
  };
  // 0) async-tensor / cluster-barrier probe (NULL descriptor: no-op)
  k_cdna5_probe<<<1, 32, 0, stream>>>();

  // 1) bf16 conversions
  cvt(x, x_bf, MROWS * D_MODEL);
  cvt(B_ssm, bssm_bf, D_MODEL * STATE);
  cvt(W_router, wrout_bf, N_COMP * D_MODEL);
  cvt(Wq, wq_bf, D_MODEL * RANK);
  cvt(Wk, wk_bf, D_MODEL * RANK);
  cvt(Wv, wv_bf, D_MODEL * RANK);
  cvt(Wo, wo_bf, D_MODEL * D_MODEL);

  // 2) xB = x @ B_ssm   (B_ssm[d,n]: sn==1, sk=STATE) — LDS-staged B
  k_gemm_bn1<false><<<dim3(64, 1), 256, 0, stream>>>(
      x_bf, D_MODEL, 0, bssm_bf, STATE, 0,
      xB, STATE, 0, 2, D_MODEL);
  // 3) router logits = x @ W_router^T  (W_router[n,d]: sk==1, sn=D) — direct
  k_gemm_bk1<false><<<dim3(64, 1), 256, 0, stream>>>(
      x_bf, D_MODEL, 0, wrout_bf, D_MODEL, 0,
      logits, N_COMP, 0, 2, D_MODEL);

  // 4) scan -> h_final -> h_proj -> importance
  k_scan<<<1, 128, 0, stream>>>(xB, A, h_final);
  k_hproj<<<(BSZ * D_MODEL + 255) / 256, 256, 0, stream>>>(h_final, W_imp, h_proj);
  k_implogit<<<(MROWS * 32 + 255) / 256, 256, 0, stream>>>(x, h_proj, il);
  k_softmaxS<<<BSZ, 256, 0, stream>>>(il, imp);

  // 5) routing weights nw
  k_zero<<<1, 128, 0, stream>>>(nwraw, BSZ * N_COMP);
  k_nw<<<(MROWS * 32 + 255) / 256, 256, 0, stream>>>(logits, imp, nwraw);
  k_nwnorm<<<BSZ, 64, 0, stream>>>(nwraw);

  // 6) shared_compress (bf16)
  k_sc<<<(D_MODEL * RANK) / 256, 256, 0, stream>>>(cn, nwraw, sc_bf);

  // 7) hcomp[b] = x[b] @ sc[b]  (sc[d,r]: sn==1, sk=RANK) — LDS-staged B
  k_gemm_bn1<true><<<dim3(64, BSZ), 256, 0, stream>>>(
      x_bf, D_MODEL, (long)SLEN * D_MODEL,
      sc_bf, RANK, (long)D_MODEL * RANK,
      hcomp_bf, RANK, (long)SLEN * RANK, 4, D_MODEL);

  // 8) Q/K/V  (M=2048, N=1024, K=128) — fully direct
  k_qkv<<<dim3(1024, BSZ), 256, 0, stream>>>(hcomp_bf, wq_bf, wk_bf, wv_bf,
                                             Qb, Kb, Vb);

  // 9) causal flash attention
  k_attn<<<dim3(SLEN / 16, BSZ * N_HEADS), 32, 0, stream>>>(Qb, Kb, Vb, attn_bf);

  // 10) output = attn_out @ Wo^T  (Wo[o,d]: sk==1, sn=D) — direct
  k_gemm_bk1<false><<<dim3(1024, 1), 256, 0, stream>>>(
      attn_bf, D_MODEL, 0, wo_bf, D_MODEL, 0,
      out, D_MODEL, 0, 32, D_MODEL);
}
